// FrustumProposer_29025388987118
// MI455X (gfx1250) — matrix operand order repaired and checked by reading, not compiled, not run on hardware
//
#include <hip/hip_runtime.h>

typedef __attribute__((ext_vector_type(16))) _Float16 v16h;
typedef __attribute__((ext_vector_type(8)))  float    v8f;

#define FP_NPTS   250000
#define FP_NDET   64
#define FP_NMAGS  10
#define FP_NROT   10
#define FP_NTRIAL 4
#define FP_FRMIN  2.0f
#define FP_FRMAX  60.0f
#define FP_BINW   5.8f
#define FP_IOUW   0.91f
#define FP_PLNW   0.92f
#define FP_THR    0.69f
#define FP_IMGW   1600.0f
#define FP_IMGH   900.0f
#define FP_PI     3.14159265358979323846f

#define FP_NCHUNK ((FP_NPTS + 31) / 32)   // 7813
#define FP_NWAVE  8                        // waves per 256-thread block

// corner template (CORNER_TEMPLATE / 1, already halved)
__device__ const float FCX[8] = { 0.5f, 0.5f,-0.5f,-0.5f, 0.5f, 0.5f,-0.5f,-0.5f};
__device__ const float FCY[8] = { 0.5f,-0.5f,-0.5f, 0.5f, 0.5f,-0.5f,-0.5f, 0.5f};
__device__ const float FCZ[8] = {-0.5f,-0.5f,-0.5f,-0.5f, 0.5f, 0.5f, 0.5f, 0.5f};

// 3x3 inverse via adjugate (matches jnp.linalg.inv numerically for well-conditioned R/K)
__device__ __forceinline__ void fp_inv3(const float* m, float* o) {
  const float a=m[0],b=m[1],c=m[2],d=m[3],e=m[4],f=m[5],g=m[6],h=m[7],i=m[8];
  const float A  =  e*i - f*h;
  const float B  = -(d*i - f*g);
  const float C  =  d*h - e*g;
  const float det = a*A + b*B + c*C;
  const float r = 1.0f / det;
  o[0]= A*r;            o[1]= -(b*i - c*h)*r; o[2]=  (b*f - c*e)*r;
  o[3]= B*r;            o[4]=  (a*i - c*g)*r; o[5]= -(a*f - c*d)*r;
  o[6]= C*r;            o[7]= -(a*h - b*g)*r; o[8]=  (a*e - b*d)*r;
}

__device__ __forceinline__ void fp_mv3(const float* m, const float* v, float* o) {
  o[0] = m[0]*v[0] + m[1]*v[1] + m[2]*v[2];
  o[1] = m[3]*v[0] + m[4]*v[1] + m[5]*v[2];
  o[2] = m[6]*v[0] + m[7]*v[1] + m[8]*v[2];
}

// ---------------------------------------------------------------------------
// Workspace layout (floats):
//   [0,512)     wsDet   : per det {dmin.xyz,0, dmax.xyz,0}   (64 x 8)
//   [512,768)   wsCdir  : per det {cdir.xyz,0}               (64 x 4)
//   [768,800)   wsPlane : per trial {n.xyz, d0, scale, pad3} (4 x 8)
//   [1024,2048) accum   : per det 16 f32 cols                (64 x 16)
//               cols 0-9 = dns counts, 10-13 = inl counts, 14 = mask sum
// ---------------------------------------------------------------------------

__global__ __launch_bounds__(128) void fp_setup_kernel(
    const float* __restrict__ points,
    const float* __restrict__ cam_intr,
    const float* __restrict__ cam2lidar,
    const float* __restrict__ img_aug,
    const float* __restrict__ lidar_aug,
    const float* __restrict__ det_boxes,
    const int*   __restrict__ det_cam_idx,
    float* __restrict__ ws)
{
  const int tid = threadIdx.x;
  float* wsDet   = ws;
  float* wsCdir  = ws + 512;
  float* wsPlane = ws + 768;
  float* accum   = ws + 1024;

  // zero the WMMA accumulators every launch (deterministic)
  for (int i = tid; i < FP_NDET * 16; i += blockDim.x) accum[i] = 0.0f;

  if (tid < FP_NDET) {
    const int d = tid;
    const int cam = det_cam_idx[d];

    const float* Km = cam_intr + cam * 16;
    float K3[9]   = {Km[0],Km[1],Km[2], Km[4],Km[5],Km[6], Km[8],Km[9],Km[10]};
    float Kinv[9]; fp_inv3(K3, Kinv);

    const float* Pm = img_aug + cam * 16;
    float PR[9]   = {Pm[0],Pm[1],Pm[2], Pm[4],Pm[5],Pm[6], Pm[8],Pm[9],Pm[10]};
    float PT[3]   = {Pm[3], Pm[7], Pm[11]};
    float PRinv[9]; fp_inv3(PR, PRinv);

    const float* Cm = cam2lidar + cam * 16;
    float CR[9]   = {Cm[0],Cm[1],Cm[2], Cm[4],Cm[5],Cm[6], Cm[8],Cm[9],Cm[10]};
    float CT[3]   = {Cm[3], Cm[7], Cm[11]};

    const float* Lm = lidar_aug;
    float ER[9]   = {Lm[0],Lm[1],Lm[2], Lm[4],Lm[5],Lm[6], Lm[8],Lm[9],Lm[10]};
    float ET[3]   = {Lm[3], Lm[7], Lm[11]};

    const float x1 = det_boxes[d*4+0], y1 = det_boxes[d*4+1];
    const float x2 = det_boxes[d*4+2], y2 = det_boxes[d*4+3];
    const float lo[3]  = {x1, y1, FP_FRMIN};
    const float hi3[3] = {x2, y2, FP_FRMAX};
    const float whl[3] = {hi3[0]-lo[0], hi3[1]-lo[1], hi3[2]-lo[2]};
    const float ctr[3] = {(hi3[0]+lo[0])*0.5f, (hi3[1]+lo[1])*0.5f, (hi3[2]+lo[2])*0.5f};

    float mn[3] = { 3.0e38f,  3.0e38f,  3.0e38f};
    float mx[3] = {-3.0e38f, -3.0e38f, -3.0e38f};
    float sum[3] = {0.0f, 0.0f, 0.0f};

    #pragma unroll
    for (int k = 0; k < 8; ++k) {
      float ci[3] = {whl[0]*FCX[k] + ctr[0], whl[1]*FCY[k] + ctr[1], whl[2]*FCZ[k] + ctr[2]};
      float a0[3] = {ci[0]-PT[0], ci[1]-PT[1], ci[2]-PT[2]};
      float b0[3]; fp_mv3(PRinv, a0, b0);
      float c0[3] = {b0[0]*b0[2], b0[1]*b0[2], b0[2]};
      float e0[3]; fp_mv3(Kinv, c0, e0);
      float f0[3]; fp_mv3(CR, e0, f0);
      f0[0]+=CT[0]; f0[1]+=CT[1]; f0[2]+=CT[2];
      float g0[3]; fp_mv3(ER, f0, g0);
      g0[0]+=ET[0]; g0[1]+=ET[1]; g0[2]+=ET[2];
      const float nm = sqrtf(g0[0]*g0[0] + g0[1]*g0[1] + g0[2]*g0[2]);
      const float ir = 1.0f / (nm + 1e-8f);
      float dir[3] = {g0[0]*ir, g0[1]*ir, g0[2]*ir};
      #pragma unroll
      for (int j = 0; j < 3; ++j) {
        mn[j] = fminf(mn[j], dir[j]);
        mx[j] = fmaxf(mx[j], dir[j]);
        sum[j] += dir[j];
      }
    }
    float md[3] = {sum[0]*0.125f, sum[1]*0.125f, sum[2]*0.125f};
    const float mg = sqrtf(md[0]*md[0] + md[1]*md[1] + md[2]*md[2]);
    const float r2 = 1.0f / (mg + 1e-8f);

    wsDet[d*8+0]=mn[0]; wsDet[d*8+1]=mn[1]; wsDet[d*8+2]=mn[2]; wsDet[d*8+3]=0.0f;
    wsDet[d*8+4]=mx[0]; wsDet[d*8+5]=mx[1]; wsDet[d*8+6]=mx[2]; wsDet[d*8+7]=0.0f;
    wsCdir[d*4+0]=md[0]*r2; wsCdir[d*4+1]=md[1]*r2; wsCdir[d*4+2]=md[2]*r2; wsCdir[d*4+3]=0.0f;
  } else if (tid < FP_NDET + FP_NTRIAL) {
    // RANSAC-style planes from fixed point triplets
    const int t = tid - FP_NDET;
    const int step = FP_NPTS / 3;                       // 83333
    const float* r0 = points + (size_t)t * 4;
    const float* r1 = points + (size_t)(t + step) * 4;
    const float* r2 = points + (size_t)(t + 2 * step) * 4;
    const float p0[3] = {r0[1], r0[2], r0[3]};
    const float u[3]  = {r1[1]-p0[0], r1[2]-p0[1], r1[3]-p0[2]};
    const float v[3]  = {r2[1]-p0[0], r2[2]-p0[1], r2[3]-p0[2]};
    const float n0 = u[1]*v[2] - u[2]*v[1];
    const float n1 = u[2]*v[0] - u[0]*v[2];
    const float n2 = u[0]*v[1] - u[1]*v[0];
    const float d0 = n0*p0[0] + n1*p0[1] + n2*p0[2];
    const float nn = n0*n0 + n1*n1 + n2*n2;
    const float sc = sqrtf(nn) / (nn + 1e-8f);          // dist = |n.x - d0| * sc
    float* pl = wsPlane + t*8;
    pl[0]=n0; pl[1]=n1; pl[2]=n2; pl[3]=d0; pl[4]=sc; pl[5]=0.f; pl[6]=0.f; pl[7]=0.f;
  }
}

// ---------------------------------------------------------------------------
// Point pass: per 32-point chunk build mask tile (A, f16 0/1) and quantity
// tile (B, f16: near[10], inl[4], 1, 0) and fuse all reductions through
// v_wmma_f32_16x16x32_f16 (4 det tiles of 16 -> 64 detections).
// ---------------------------------------------------------------------------
__global__ __launch_bounds__(256) void fp_point_kernel(
    const float* __restrict__ points,
    float* __restrict__ ws)
{
  __shared__ float    s_pdir[FP_NWAVE][32][4];   // dir.xyz, (valid ? |p| : -1)
  __shared__ _Float16 s_qty [FP_NWAVE][32][16];  // B-tile source, per point
  __shared__ unsigned s_msk [FP_NWAVE][64];      // per-det 32-point bitmask
  __shared__ float    s_acc [FP_NDET * 16];      // block-level f32 accumulator

  const float* wsDet = ws;
  const float* wsPl  = ws + 768;
  float* accum       = ws + 1024;

  const int tid  = threadIdx.x;
  const int lane = tid & 31;
  const int w    = tid >> 5;
  const int hi   = lane >> 4;      // half-wave select
  const int col  = lane & 15;
  const int waveId = blockIdx.x * FP_NWAVE + w;
  const int nWaves = gridDim.x * FP_NWAVE;

  for (int i = tid; i < FP_NDET * 16; i += blockDim.x) s_acc[i] = 0.0f;
  __syncthreads();

  // per-lane frustum bounds: detections lane and lane+32
  const float4 dmn0 = *(const float4*)(wsDet + lane * 8);
  const float4 dmx0 = *(const float4*)(wsDet + lane * 8 + 4);
  const float4 dmn1 = *(const float4*)(wsDet + (lane + 32) * 8);
  const float4 dmx1 = *(const float4*)(wsDet + (lane + 32) * 8 + 4);

  float pnx[4], pny[4], pnz[4], pd0[4], psc[4];
  #pragma unroll
  for (int t = 0; t < 4; ++t) {
    const float* pl = wsPl + t * 8;
    pnx[t]=pl[0]; pny[t]=pl[1]; pnz[t]=pl[2]; pd0[t]=pl[3]; psc[t]=pl[4];
  }
  float mags[FP_NMAGS];
  #pragma unroll
  for (int m = 0; m < FP_NMAGS; ++m)
    mags[m] = FP_FRMIN + (float)m * ((FP_FRMAX - FP_FRMIN) / 9.0f);

  v8f acc[4] = {};

  for (int chunk = waveId; chunk < FP_NCHUNK; chunk += nWaves) {
    const int idx  = chunk * 32 + lane;
    const int ridx = (idx < FP_NPTS) ? idx : FP_NPTS - 1;

    // prefetch next chunk (global_prefetch_b8)
    {
      long ni = (long)(chunk + nWaves) * 32 + lane;
      if (ni >= FP_NPTS) ni = FP_NPTS - 1;
      __builtin_prefetch(points + ni * 4, 0, 3);
    }

    const float4 pt = ((const float4*)points)[ridx];     // [intensity, x, y, z]
    const float px = pt.y, py = pt.z, pz = pt.w;
    const float pm = sqrtf(px*px + py*py + pz*pz);
    const float ir = 1.0f / (pm + 1e-8f);
    s_pdir[w][lane][0] = px * ir;
    s_pdir[w][lane][1] = py * ir;
    s_pdir[w][lane][2] = pz * ir;
    s_pdir[w][lane][3] = (idx < FP_NPTS) ? pm : -1.0f;   // sentinel kills mask

    #pragma unroll
    for (int m = 0; m < FP_NMAGS; ++m)
      s_qty[w][lane][m] = (fabsf(pm - mags[m]) < FP_BINW) ? (_Float16)1.0f : (_Float16)0.0f;
    #pragma unroll
    for (int t = 0; t < FP_NTRIAL; ++t) {
      const float dist = fabsf(px*pnx[t] + py*pny[t] + pz*pnz[t] - pd0[t]) * psc[t];
      s_qty[w][lane][10 + t] = (dist < FP_THR) ? (_Float16)1.0f : (_Float16)0.0f;
    }
    s_qty[w][lane][14] = (_Float16)1.0f;                 // mask-sum column
    s_qty[w][lane][15] = (_Float16)0.0f;

    asm volatile("s_wait_dscnt 0x0" ::: "memory");       // wave-local LDS visibility

    // each lane builds the 32-point mask word for dets lane and lane+32
    unsigned m0 = 0u, m1 = 0u;
    #pragma unroll 4
    for (int k = 0; k < 32; ++k) {
      const float qx = s_pdir[w][k][0];
      const float qy = s_pdir[w][k][1];
      const float qz = s_pdir[w][k][2];
      const float qm = s_pdir[w][k][3];
      const bool rng = (qm >= FP_FRMIN) & (qm <= FP_FRMAX);
      const bool b0 = rng & (qx>=dmn0.x) & (qx<=dmx0.x) & (qy>=dmn0.y) & (qy<=dmx0.y)
                          & (qz>=dmn0.z) & (qz<=dmx0.z);
      const bool b1 = rng & (qx>=dmn1.x) & (qx<=dmx1.x) & (qy>=dmn1.y) & (qy<=dmx1.y)
                          & (qz>=dmn1.z) & (qz<=dmx1.z);
      m0 |= (b0 ? 1u : 0u) << k;
      m1 |= (b1 ? 1u : 0u) << k;
    }
    s_msk[w][lane]      = m0;
    s_msk[w][lane + 32] = m1;

    asm volatile("s_wait_dscnt 0x0" ::: "memory");

    // B fragment: 32x16, column N = lane&15; lanes 0-15 hold K0-15,
    // lanes 16-31 hold K16-31 (contiguous k order, same as A)
    v16h b;
    #pragma unroll
    for (int e = 0; e < 16; ++e) {
      const int kb = e + (hi ? 16 : 0);
      b[e] = s_qty[w][kb][col];
    }

    // A fragments: 16x32, row M = lane&15; element->K contiguous (matches B):
    // lanes 0-15: K = 0..15 ; lanes 16-31: K = 16..31
    #pragma unroll
    for (int t = 0; t < 4; ++t) {
      const unsigned mw = s_msk[w][t * 16 + col];
      v16h a;
      #pragma unroll
      for (int e = 0; e < 16; ++e) {
        const int ka = e + (hi ? 16 : 0);
        a[e] = __builtin_bit_cast(_Float16,
                 (unsigned short)(((mw >> ka) & 1u) ? 0x3C00u : 0u));
      }
      acc[t] = __builtin_amdgcn_wmma_f32_16x16x32_f16(
                 false, a, false, b, (short)0, acc[t], false, false);
    }
    asm volatile("s_wait_dscnt 0x0" ::: "memory");       // reads done before next-iter stores
  }

  // C/D layout: lane L holds N=L&15; VGPR r holds M=r (lanes 0-15) / r+8 (16-31)
  // Block-level LDS reduction (ds_add_f32), then one global f32 atomic per entry.
  // All values are integer-valued -> order-independent, bit-exact.
  #pragma unroll
  for (int t = 0; t < 4; ++t) {
    #pragma unroll
    for (int r = 0; r < 8; ++r) {
      const int det = t * 16 + r + 8 * hi;
      atomicAdd(&s_acc[det * 16 + col], acc[t][r]);
    }
  }
  __syncthreads();
  for (int i = tid; i < FP_NDET * 16; i += blockDim.x)
    atomicAdd(accum + i, s_acc[i]);
}

// ---------------------------------------------------------------------------
// Scoring: one block per detection, one thread per (mag, rot) candidate.
// ---------------------------------------------------------------------------
__global__ __launch_bounds__(128) void fp_score_kernel(
    const float* __restrict__ cam_intr,
    const float* __restrict__ cam2lidar,
    const float* __restrict__ img_aug,
    const float* __restrict__ lidar_aug,
    const float* __restrict__ det_boxes,
    const float* __restrict__ det_scores,
    const float* __restrict__ anchors,
    const int*   __restrict__ det_labels,
    const int*   __restrict__ det_cam_idx,
    const float* __restrict__ ws,
    float* __restrict__ out)
{
  __shared__ float ssc[128];
  __shared__ int   sid[128];

  const int d   = blockIdx.x;
  const int tid = threadIdx.x;
  const float* wsCdir = ws + 512;
  const float* accum  = ws + 1024;

  const int cam = det_cam_idx[d];
  const int lbl = det_labels[d];
  const float x1 = det_boxes[d*4+0], y1 = det_boxes[d*4+1];
  const float x2 = det_boxes[d*4+2], y2 = det_boxes[d*4+3];
  const float cd0 = wsCdir[d*4+0], cd1 = wsCdir[d*4+1], cd2 = wsCdir[d*4+2];
  const float sx = anchors[lbl*3+0], sy = anchors[lbl*3+1], sz = anchors[lbl*3+2];

  float score = -3.0e38f;
  int   myidx = 0x7FFFFFFF;

  if (tid < FP_NMAGS * FP_NROT) {
    const int m = tid / FP_NROT;
    const int r = tid % FP_NROT;

    const float* Km = cam_intr + cam * 16;
    float K3[9] = {Km[0],Km[1],Km[2], Km[4],Km[5],Km[6], Km[8],Km[9],Km[10]};
    const float* Pm = img_aug + cam * 16;
    float PR[9] = {Pm[0],Pm[1],Pm[2], Pm[4],Pm[5],Pm[6], Pm[8],Pm[9],Pm[10]};
    float PT[3] = {Pm[3], Pm[7], Pm[11]};
    const float* Cm = cam2lidar + cam * 16;
    float CR[9] = {Cm[0],Cm[1],Cm[2], Cm[4],Cm[5],Cm[6], Cm[8],Cm[9],Cm[10]};
    float CT[3] = {Cm[3], Cm[7], Cm[11]};
    float CRinv[9]; fp_inv3(CR, CRinv);
    const float* Lm = lidar_aug;
    float ER[9] = {Lm[0],Lm[1],Lm[2], Lm[4],Lm[5],Lm[6], Lm[8],Lm[9],Lm[10]};
    float ET[3] = {Lm[3], Lm[7], Lm[11]};
    float ERinv[9]; fp_inv3(ER, ERinv);

    const float mag = FP_FRMIN + (float)m * ((FP_FRMAX - FP_FRMIN) / 9.0f);
    const float ry  = (float)r * (FP_PI / 9.0f);
    const float cc = cosf(ry), sn = sinf(ry);
    const float ctr0 = cd0 * mag, ctr1 = cd1 * mag, ctr2 = cd2 * mag;

    float bx1 = 3.0e38f, by1 = 3.0e38f, bx2 = -3.0e38f, by2 = -3.0e38f;
    #pragma unroll
    for (int k = 0; k < 8; ++k) {
      const float lx = sx * FCX[k], ly = sy * FCY[k], lz = sz * FCZ[k];
      const float wx = lx*cc - ly*sn + ctr0;
      const float wy = lx*sn + ly*cc + ctr1;
      const float wz = lz + ctr2;
      float a0[3] = {wx - ET[0], wy - ET[1], wz - ET[2]};
      float b0[3]; fp_mv3(ERinv, a0, b0);
      float c0[3] = {b0[0]-CT[0], b0[1]-CT[1], b0[2]-CT[2]};
      float e0[3]; fp_mv3(CRinv, c0, e0);
      float q0[3]; fp_mv3(K3, e0, q0);
      const float depth = fmaxf(q0[2], 0.001f);
      float qq[3] = {q0[0]/depth, q0[1]/depth, depth};
      float f0[3]; fp_mv3(PR, qq, f0);
      float u = fminf(fmaxf(f0[0] + PT[0], 0.0f), FP_IMGW);
      float v = fminf(fmaxf(f0[1] + PT[1], 0.0f), FP_IMGH);
      bx1 = fminf(bx1, u); by1 = fminf(by1, v);
      bx2 = fmaxf(bx2, u); by2 = fmaxf(by2, v);
    }
    const float ix1 = fmaxf(bx1, x1), iy1 = fmaxf(by1, y1);
    const float ix2 = fminf(bx2, x2), iy2 = fminf(by2, y2);
    const float inter  = fmaxf(ix2 - ix1, 0.0f) * fmaxf(iy2 - iy1, 0.0f);
    const float area_p = (bx2 - bx1) * (by2 - by1);
    const float area_d = (x2 - x1) * (y2 - y1);
    const float iou = inter / (area_p + area_d - inter + 1e-6f);

    float dmx = accum[d*16 + 0];
    #pragma unroll
    for (int mm = 1; mm < FP_NMAGS; ++mm) dmx = fmaxf(dmx, accum[d*16 + mm]);
    const float dns = accum[d*16 + m] / (dmx + 1e-6f);

    const float msum = accum[d*16 + 14] + 1e-6f;
    float pln = accum[d*16 + 10] / msum;
    #pragma unroll
    for (int t = 1; t < FP_NTRIAL; ++t) pln = fmaxf(pln, accum[d*16 + 10 + t] / msum);

    score = FP_PLNW * (FP_IOUW * iou + (1.0f - FP_IOUW) * dns)
          + (1.0f - FP_PLNW) * pln;
    myidx = m * FP_NROT + r;
  }

  ssc[tid] = score; sid[tid] = myidx;
  __syncthreads();
  for (int off = 64; off > 0; off >>= 1) {
    if (tid < off) {
      const float s2 = ssc[tid + off];
      const int   i2 = sid[tid + off];
      if (s2 > ssc[tid] || (s2 == ssc[tid] && i2 < sid[tid])) { ssc[tid] = s2; sid[tid] = i2; }
    }
    __syncthreads();
  }
  if (tid == 0) {
    const int best = sid[0];
    const int bm = best / FP_NROT, br = best % FP_NROT;
    const float mag = FP_FRMIN + (float)bm * ((FP_FRMAX - FP_FRMIN) / 9.0f);
    out[d*8+0] = cd0 * mag;
    out[d*8+1] = cd1 * mag;
    out[d*8+2] = cd2 * mag;
    out[d*8+3] = sx;
    out[d*8+4] = sy;
    out[d*8+5] = sz;
    out[d*8+6] = (float)br * (FP_PI / 9.0f);
    out[d*8+7] = ssc[0] * det_scores[d];
  }
}

extern "C" void kernel_launch(void* const* d_in, const int* in_sizes, int n_in,
                              void* d_out, int out_size, void* d_ws, size_t ws_size,
                              hipStream_t stream) {
  (void)in_sizes; (void)n_in; (void)out_size; (void)ws_size;
  const float* points      = (const float*)d_in[0];
  const float* cam_intr    = (const float*)d_in[1];
  const float* cam2lidar   = (const float*)d_in[2];
  const float* img_aug     = (const float*)d_in[3];
  const float* lidar_aug   = (const float*)d_in[4];
  const float* det_boxes   = (const float*)d_in[5];
  const float* det_scores  = (const float*)d_in[6];
  const float* anchors     = (const float*)d_in[7];
  const int*   det_labels  = (const int*)d_in[8];
  const int*   det_cam_idx = (const int*)d_in[9];
  float* out = (float*)d_out;
  float* ws  = (float*)d_ws;

  fp_setup_kernel<<<1, 128, 0, stream>>>(points, cam_intr, cam2lidar, img_aug,
                                         lidar_aug, det_boxes, det_cam_idx, ws);
  fp_point_kernel<<<256, 256, 0, stream>>>(points, ws);
  fp_score_kernel<<<FP_NDET, 128, 0, stream>>>(cam_intr, cam2lidar, img_aug,
                                               lidar_aug, det_boxes, det_scores,
                                               anchors, det_labels, det_cam_idx,
                                               ws, out);
}
